// E2EGanImputationGenerator_36378372997187
// MI455X (gfx1250) — matrix operand order repaired and checked by reading, not compiled, not run on hardware
//
#include <hip/hip_runtime.h>

// ---------------------------------------------------------------------------
// GRU-I encoder/decoder (GAN imputation generator) for gfx1250 / MI455X.
// bf16 WMMA (v_wmma_f32_16x16x32_bf16) GEMMs with fused gate epilogues and
// async global->LDS (ASYNCcnt) double-buffered weight staging.
// ---------------------------------------------------------------------------

#define B_DIM 1024
#define T_DIM 96
#define I_DIM 256
#define H_DIM 1024
#define AW    1280   // activation row width: [h (1024) | x_t (256)]

typedef __bf16 bf16_t;
typedef float  v8f   __attribute__((ext_vector_type(8)));
typedef __bf16 v16bf __attribute__((ext_vector_type(16)));
typedef unsigned int u32x4 __attribute__((ext_vector_type(4)));

union Frag { u32x4 q[2]; v16bf v; };

struct GemmParams {
  const bf16_t* A;      // activation matrix (bf16, row-major)
  int lda;              // row stride in elements
  int K;                // reduction depth (multiple of 32)
  int N;                // output columns
  const bf16_t* Bp;     // packed weights (fragment-tiled)
  const float* bias0;   // bias for n < 1024 (or all)
  const float* bias1;   // bias for n >= 1024 (GATES mode)
  float* hbuf;          // f32 hidden / h~ buffer [B,H]
  float* mubuf;         // f32 mu buffer [B,H]
  bf16_t* abuf;         // activation buffer [B,AW]
  bf16_t* abuf2;        // second activation buffer [B,AW]
  bf16_t* zbuf;         // z bf16 [B,H]
  float* out;           // final output [B,T,I]
  int t;                // timestep (REG mode)
};

enum { M_BETA = 0, M_GATES = 1, M_HHAT = 2, M_Z = 3, M_DI = 4, M_REG = 5 };

__device__ __forceinline__ float sigmoidf_(float v) {
  return 1.0f / (1.0f + __expf(-v));
}

// Async DMA of one 16B chunk into LDS (per lane). LDS byte offset comes from
// the low 32 bits of the generic LDS address (flat aperture maps addr[31:0]
// directly onto the LDS offset). Tracked by ASYNCcnt.
__device__ __forceinline__ void async_ld_b128(unsigned lds_off, const void* gptr) {
  asm volatile("global_load_async_to_lds_b128 %0, %1, off"
               :: "v"(lds_off), "v"((unsigned long long)(uintptr_t)gptr)
               : "memory");
}
__device__ __forceinline__ void wait_asynccnt0() {
  asm volatile("s_wait_asynccnt 0x0" ::: "memory");
}

// ---------------------------------------------------------------------------
// Fused WMMA GEMM: C = A[B_DIM,K] * Bp[K,N] (+bias, epilogue per MODE)
// Block: 256 threads (8 wave32s). Tile: 128 rows x 64 cols.
// Wave w owns rows [tileM + 16w, +16), 4 accumulators of 16 cols each.
// Packed-B layout: tile(kt,nt) is 32x16 (KxN) stored lane-major:
//   element(lane, e) = W[kt*32 + (lane>>4)*16 + e][nt*16 + (lane&15)]
// so a lane's whole fragment is 32 contiguous bytes.
// B tiles are DMA'd global->LDS asynchronously, double buffered: buffer
// (kt+1)&1 is filled while buffer kt&1 feeds the 4 WMMAs.
// ---------------------------------------------------------------------------
template <int MODE>
__global__ __launch_bounds__(256) void gemm_cell(GemmParams p) {
  __shared__ __align__(16) bf16_t ldsB[2 * 4 * 512];  // 2 buffers x 4 subtiles x 1KB

  const int tid   = threadIdx.x;
  const int wave  = tid >> 5;
  const int lane  = tid & 31;
  const int tileN = blockIdx.x * 64;
  const int tileM = blockIdx.y * 128;
  const int NT     = p.N >> 4;      // total 16-col tiles
  const int ntBase = tileN >> 4;
  const int row0   = tileM + wave * 16;

  // A fragment addressing (contiguous k order, same as packed B):
  //   lane l: m = l&15, 16 contiguous elems at K = koff..koff+15, koff=(l>>4)*16
  const int m    = row0 + (lane & 15);
  const int koff = (lane >> 4) << 4;
  // B staging: 256 threads x 16B = 4KB per K-step
  const int jj = tid >> 6;
  const int cc = tid & 63;
  const unsigned ldsOff0   = (unsigned)(uintptr_t)(&ldsB[jj * 512 + cc * 8]);
  const unsigned ldsStride = 4 * 512 * (unsigned)sizeof(bf16_t);  // 4KB

  v8f acc[4];
#pragma unroll
  for (int j = 0; j < 4; ++j)
    acc[j] = (v8f){0.f, 0.f, 0.f, 0.f, 0.f, 0.f, 0.f, 0.f};

  const int nkt = p.K >> 5;

  // Prologue: kick off DMA of K-tile 0 into buffer 0.
  async_ld_b128(ldsOff0, p.Bp + (size_t)(ntBase + jj) * 512 + (size_t)cc * 8);

  for (int kt = 0; kt < nkt; ++kt) {
    wait_asynccnt0();     // this wave's DMA for buffer kt&1 has landed in LDS
    __syncthreads();      // everyone's DMA landed; prior reads of other buffer done

    if (kt + 1 < nkt) {   // overlap next tile's DMA with this tile's math
      const bf16_t* g =
          p.Bp + ((size_t)(kt + 1) * NT + (size_t)(ntBase + jj)) * 512 + (size_t)cc * 8;
      async_ld_b128(ldsOff0 + (((kt + 1) & 1) ? ldsStride : 0u), g);
    }

    const bf16_t* arow = p.A + (size_t)m * (size_t)p.lda + (size_t)(kt << 5);
    Frag a;
    a.q[0] = *(const u32x4*)(arow + koff);
    a.q[1] = *(const u32x4*)(arow + koff + 8);

    const bf16_t* lb = &ldsB[(kt & 1) * (4 * 512)];
    Frag bfr[4];
#pragma unroll
    for (int j = 0; j < 4; ++j) {
      bfr[j].q[0] = *(const u32x4*)(&lb[j * 512 + lane * 16]);
      bfr[j].q[1] = *(const u32x4*)(&lb[j * 512 + lane * 16 + 8]);
    }
#pragma unroll
    for (int j = 0; j < 4; ++j) {
      acc[j] = __builtin_amdgcn_wmma_f32_16x16x32_bf16(
          false, a.v, false, bfr[j].v, (short)0, acc[j], false, false);
    }
  }

  // Epilogue. C/D layout: VGPR r -> lanes 0-15: M=r, lanes 16-31: M=r+8.
  const int rbase = row0 + ((lane >> 4) << 3);
  const int nlo   = lane & 15;
#pragma unroll
  for (int j = 0; j < 4; ++j) {
    const int n = tileN + j * 16 + nlo;
#pragma unroll
    for (int r = 0; r < 8; ++r) {
      const int b    = rbase + r;
      const float av = acc[j][r];

      if constexpr (MODE == M_BETA) {
        // beta = exp(-relu(delta@Wtd+b)); h~ = beta*h
        const float v    = av + p.bias0[n];
        const float beta = __expf(-fmaxf(v, 0.0f));
        const size_t hi  = (size_t)b * H_DIM + n;
        const float hb   = beta * p.hbuf[hi];
        p.hbuf[hi]                 = hb;
        p.abuf[(size_t)b * AW + n] = (bf16_t)hb;
      } else if constexpr (MODE == M_GATES) {
        if (n < H_DIM) {
          const float rr = sigmoidf_(av + p.bias0[n]);
          const float rh = rr * p.hbuf[(size_t)b * H_DIM + n];
          p.abuf2[(size_t)b * AW + n] = (bf16_t)rh;
        } else {
          const int   nn = n - H_DIM;
          const float mu = sigmoidf_(av + p.bias1[nn]);
          p.mubuf[(size_t)b * H_DIM + nn] = mu;
        }
      } else if constexpr (MODE == M_HHAT) {
        const float hh  = tanhf(av + p.bias0[n]);
        const size_t hi = (size_t)b * H_DIM + n;
        const float mu  = p.mubuf[hi];
        const float hb  = p.hbuf[hi];
        const float hn  = (1.0f - mu) * hb + mu * hh;
        p.hbuf[hi]                 = hn;
        p.abuf[(size_t)b * AW + n] = (bf16_t)hn;
      } else if constexpr (MODE == M_Z) {
        p.zbuf[(size_t)b * H_DIM + n] = (bf16_t)(av + p.bias0[n]);
      } else if constexpr (MODE == M_DI) {
        const bf16_t v = (bf16_t)(av + p.bias0[n]);
        p.abuf[(size_t)b * AW + H_DIM + n]  = v;
        p.abuf2[(size_t)b * AW + H_DIM + n] = v;
      } else if constexpr (MODE == M_REG) {
        const float xn = tanhf(av + p.bias0[n]);
        p.out[(size_t)b * (T_DIM * I_DIM) + (size_t)p.t * I_DIM + n] = xn;
        const bf16_t v = (bf16_t)xn;
        p.abuf[(size_t)b * AW + H_DIM + n]  = v;
        p.abuf2[(size_t)b * AW + H_DIM + n] = v;
      }
    }
  }
}

// ---------------------------------------------------------------------------
// Weight packing: f32 [K,N] -> bf16 fragment tiles (see layout above).
// ---------------------------------------------------------------------------
__global__ __launch_bounds__(256) void pack_w_kernel(const float* __restrict__ src,
                                                     bf16_t* __restrict__ dst,
                                                     int K, int N, int NTtot, int ntOff) {
  const size_t idx = (size_t)blockIdx.x * 256 + threadIdx.x;
  if (idx >= (size_t)K * N) return;
  const int k = (int)(idx / N);
  const int n = (int)(idx % N);
  const int kt = k >> 5, ko = k & 31;
  const int nt = n >> 4, nl = n & 15;
  const int lane = ((ko >> 4) << 4) | nl;
  const int e    = ko & 15;
  dst[((size_t)kt * NTtot + (size_t)(ntOff + nt)) * 512 + lane * 16 + e] = (bf16_t)src[idx];
}

// Per-timestep input prep: x_t = x*mask+noise into both activation buffers,
// delta_t into dbuf (all bf16). Encoder variant also writes x; decoder only d.
__global__ __launch_bounds__(256) void prep_enc_kernel(const float* __restrict__ x,
                                                       const float* __restrict__ mask,
                                                       const float* __restrict__ noise,
                                                       const float* __restrict__ delta,
                                                       bf16_t* abuf, bf16_t* abuf2,
                                                       bf16_t* dbuf, int t) {
  const int idx = blockIdx.x * 256 + threadIdx.x;
  if (idx >= B_DIM * I_DIM) return;
  const int b = idx >> 8;
  const int i = idx & 255;
  const size_t s = (size_t)b * (T_DIM * I_DIM) + (size_t)t * I_DIM + i;
  const bf16_t xb = (bf16_t)(x[s] * mask[s] + noise[s]);
  abuf[(size_t)b * AW + H_DIM + i]  = xb;
  abuf2[(size_t)b * AW + H_DIM + i] = xb;
  dbuf[idx] = (bf16_t)delta[s];
}

__global__ __launch_bounds__(256) void prep_dec_kernel(const float* __restrict__ delta,
                                                       bf16_t* dbuf, int t) {
  const int idx = blockIdx.x * 256 + threadIdx.x;
  if (idx >= B_DIM * I_DIM) return;
  const int b = idx >> 8;
  const int i = idx & 255;
  dbuf[idx] = (bf16_t)delta[(size_t)b * (T_DIM * I_DIM) + (size_t)t * I_DIM + i];
}

__global__ __launch_bounds__(256) void zero_f32_kernel(float* p, int n) {
  const int idx = blockIdx.x * 256 + threadIdx.x;
  if (idx < n) p[idx] = 0.0f;
}

// ---------------------------------------------------------------------------
static void run_gemm(int mode, const GemmParams& p, hipStream_t s) {
  dim3 g(p.N / 64, B_DIM / 128), b(256);
  switch (mode) {
    case M_BETA:  gemm_cell<M_BETA><<<g, b, 0, s>>>(p);  break;
    case M_GATES: gemm_cell<M_GATES><<<g, b, 0, s>>>(p); break;
    case M_HHAT:  gemm_cell<M_HHAT><<<g, b, 0, s>>>(p);  break;
    case M_Z:     gemm_cell<M_Z><<<g, b, 0, s>>>(p);     break;
    case M_DI:    gemm_cell<M_DI><<<g, b, 0, s>>>(p);    break;
    case M_REG:   gemm_cell<M_REG><<<g, b, 0, s>>>(p);   break;
  }
}

extern "C" void kernel_launch(void* const* d_in, const int* in_sizes, int n_in,
                              void* d_out, int out_size, void* d_ws, size_t ws_size,
                              hipStream_t stream) {
  (void)in_sizes; (void)n_in; (void)out_size; (void)ws_size;

  const float* x       = (const float*)d_in[0];
  const float* mask    = (const float*)d_in[1];
  const float* delta   = (const float*)d_in[2];
  const float* noise   = (const float*)d_in[3];
  const float* enc_Wtd = (const float*)d_in[4];
  const float* enc_btd = (const float*)d_in[5];
  const float* enc_Wr  = (const float*)d_in[6];
  const float* enc_br  = (const float*)d_in[7];
  const float* enc_Wmu = (const float*)d_in[8];
  const float* enc_bmu = (const float*)d_in[9];
  const float* enc_Whh = (const float*)d_in[10];
  const float* enc_bhh = (const float*)d_in[11];
  const float* z_W     = (const float*)d_in[12];
  const float* z_b     = (const float*)d_in[13];
  const float* di_W    = (const float*)d_in[14];
  const float* di_b    = (const float*)d_in[15];
  const float* dec_Wtd = (const float*)d_in[16];
  const float* dec_btd = (const float*)d_in[17];
  const float* dec_Wr  = (const float*)d_in[18];
  const float* dec_br  = (const float*)d_in[19];
  const float* dec_Wmu = (const float*)d_in[20];
  const float* dec_bmu = (const float*)d_in[21];
  const float* dec_Whh = (const float*)d_in[22];
  const float* dec_bhh = (const float*)d_in[23];
  const float* reg_W   = (const float*)d_in[24];
  const float* reg_b   = (const float*)d_in[25];
  float* out = (float*)d_out;

  // ---- workspace carve-up (all offsets 256B aligned) ----
  char* ws = (char*)d_ws;
  size_t off = 0;
  auto carve = [&](size_t bytes) -> void* {
    void* p = ws + off;
    off += (bytes + 255) & ~(size_t)255;
    return p;
  };
  bf16_t* enc_Wtd_p  = (bf16_t*)carve((size_t)I_DIM * H_DIM * 2);
  bf16_t* enc_Wrmu_p = (bf16_t*)carve((size_t)AW * 2 * H_DIM * 2);
  bf16_t* enc_Whh_p  = (bf16_t*)carve((size_t)AW * H_DIM * 2);
  bf16_t* dec_Wtd_p  = (bf16_t*)carve((size_t)I_DIM * H_DIM * 2);
  bf16_t* dec_Wrmu_p = (bf16_t*)carve((size_t)AW * 2 * H_DIM * 2);
  bf16_t* dec_Whh_p  = (bf16_t*)carve((size_t)AW * H_DIM * 2);
  bf16_t* zW_p       = (bf16_t*)carve((size_t)H_DIM * H_DIM * 2);
  bf16_t* diW_p      = (bf16_t*)carve((size_t)H_DIM * I_DIM * 2);
  bf16_t* regW_p     = (bf16_t*)carve((size_t)H_DIM * I_DIM * 2);
  bf16_t* abuf       = (bf16_t*)carve((size_t)B_DIM * AW * 2);
  bf16_t* abuf2      = (bf16_t*)carve((size_t)B_DIM * AW * 2);
  bf16_t* dbuf       = (bf16_t*)carve((size_t)B_DIM * I_DIM * 2);
  bf16_t* zbuf       = (bf16_t*)carve((size_t)B_DIM * H_DIM * 2);
  float*  hbuf       = (float*)carve((size_t)B_DIM * H_DIM * 4);
  float*  mubuf      = (float*)carve((size_t)B_DIM * H_DIM * 4);

  // ---- pack weights to bf16 fragment tiles ----
  auto pack = [&](const float* src, bf16_t* dst, int K, int N, int NTtot, int ntOff) {
    const size_t tot = (size_t)K * N;
    pack_w_kernel<<<(unsigned)((tot + 255) / 256), 256, 0, stream>>>(src, dst, K, N, NTtot, ntOff);
  };
  pack(enc_Wtd, enc_Wtd_p, I_DIM, H_DIM, H_DIM / 16, 0);
  pack(enc_Wr,  enc_Wrmu_p, AW, H_DIM, 2 * H_DIM / 16, 0);
  pack(enc_Wmu, enc_Wrmu_p, AW, H_DIM, 2 * H_DIM / 16, H_DIM / 16);
  pack(enc_Whh, enc_Whh_p, AW, H_DIM, H_DIM / 16, 0);
  pack(dec_Wtd, dec_Wtd_p, I_DIM, H_DIM, H_DIM / 16, 0);
  pack(dec_Wr,  dec_Wrmu_p, AW, H_DIM, 2 * H_DIM / 16, 0);
  pack(dec_Wmu, dec_Wrmu_p, AW, H_DIM, 2 * H_DIM / 16, H_DIM / 16);
  pack(dec_Whh, dec_Whh_p, AW, H_DIM, H_DIM / 16, 0);
  pack(z_W,  zW_p,  H_DIM, H_DIM, H_DIM / 16, 0);
  pack(di_W, diW_p, H_DIM, I_DIM, I_DIM / 16, 0);
  pack(reg_W, regW_p, H_DIM, I_DIM, I_DIM / 16, 0);

  const int nBH = B_DIM * H_DIM;
  zero_f32_kernel<<<(nBH + 255) / 256, 256, 0, stream>>>(hbuf, nBH);

  GemmParams p{};
  p.hbuf = hbuf; p.mubuf = mubuf; p.abuf = abuf; p.abuf2 = abuf2;
  p.zbuf = zbuf; p.out = out;

  // ---- encoder scan ----
  for (int t = 0; t < T_DIM; ++t) {
    prep_enc_kernel<<<(B_DIM * I_DIM + 255) / 256, 256, 0, stream>>>(
        x, mask, noise, delta, abuf, abuf2, dbuf, t);

    p.A = dbuf; p.lda = I_DIM; p.K = I_DIM; p.N = H_DIM;
    p.Bp = enc_Wtd_p; p.bias0 = enc_btd; p.bias1 = nullptr;
    run_gemm(M_BETA, p, stream);

    p.A = abuf; p.lda = AW; p.K = AW; p.N = 2 * H_DIM;
    p.Bp = enc_Wrmu_p; p.bias0 = enc_br; p.bias1 = enc_bmu;
    run_gemm(M_GATES, p, stream);

    p.A = abuf2; p.lda = AW; p.K = AW; p.N = H_DIM;
    p.Bp = enc_Whh_p; p.bias0 = enc_bhh; p.bias1 = nullptr;
    run_gemm(M_HHAT, p, stream);
  }

  // ---- latent + decoder input ----
  p.A = abuf; p.lda = AW; p.K = H_DIM; p.N = H_DIM;
  p.Bp = zW_p; p.bias0 = z_b; p.bias1 = nullptr;
  run_gemm(M_Z, p, stream);

  p.A = zbuf; p.lda = H_DIM; p.K = H_DIM; p.N = I_DIM;
  p.Bp = diW_p; p.bias0 = di_b; p.bias1 = nullptr;
  run_gemm(M_DI, p, stream);

  zero_f32_kernel<<<(nBH + 255) / 256, 256, 0, stream>>>(hbuf, nBH);

  // ---- decoder scan ----
  for (int t = 0; t < T_DIM; ++t) {
    prep_dec_kernel<<<(B_DIM * I_DIM + 255) / 256, 256, 0, stream>>>(delta, dbuf, t);

    p.A = dbuf; p.lda = I_DIM; p.K = I_DIM; p.N = H_DIM;
    p.Bp = dec_Wtd_p; p.bias0 = dec_btd; p.bias1 = nullptr;
    run_gemm(M_BETA, p, stream);

    p.A = abuf; p.lda = AW; p.K = AW; p.N = 2 * H_DIM;
    p.Bp = dec_Wrmu_p; p.bias0 = dec_br; p.bias1 = dec_bmu;
    run_gemm(M_GATES, p, stream);

    p.A = abuf2; p.lda = AW; p.K = AW; p.N = H_DIM;
    p.Bp = dec_Whh_p; p.bias0 = dec_bhh; p.bias1 = nullptr;
    run_gemm(M_HHAT, p, stream);

    p.A = abuf; p.lda = AW; p.K = H_DIM; p.N = I_DIM;
    p.Bp = regW_p; p.bias0 = reg_b; p.bias1 = nullptr;
    p.t = t;
    run_gemm(M_REG, p, stream);
  }
}
